// MultiAttentionBlock_65343632441736
// MI455X (gfx1250) — matrix operand order repaired and checked by reading, not compiled, not run on hardware
//
#include <hip/hip_runtime.h>
#include <cstdint>

#define HEADS 6
#define C_DIM 96
#define HD    16
#define B_DIM 2
#define H_DIM 256
#define W_DIM 256
#define HW    65536
#define NS    131072   /* B*HW : pixel count / stride of internal matrices */
#define EPS_LN 1e-5f
#define HID   255
#define FFN_M 510
#define FFN_MP 512
#define FFN_KP 256

typedef __attribute__((ext_vector_type(16))) _Float16 v16h;
typedef __attribute__((ext_vector_type(8)))  _Float16 v8h;
typedef __attribute__((ext_vector_type(8)))  float    v8f;

// ---------------- layout conversion ----------------
__global__ void k_cvt_nchw_to_cn(const float* __restrict__ src, float* __restrict__ dst) {
    size_t i = (size_t)blockIdx.x * blockDim.x + threadIdx.x;       // < B*C*HW
    int b  = (int)(i / ((size_t)C_DIM * HW));
    int rm = (int)(i % ((size_t)C_DIM * HW));
    int c  = rm / HW;
    int hw = rm % HW;
    dst[(size_t)c * NS + (size_t)b * HW + hw] = src[i];
}

__global__ void k_cn_to_nchw(const float* __restrict__ src, float* __restrict__ dst) {
    size_t i = (size_t)blockIdx.x * blockDim.x + threadIdx.x;
    int b  = (int)(i / ((size_t)C_DIM * HW));
    int rm = (int)(i % ((size_t)C_DIM * HW));
    int c  = rm / HW;
    int hw = rm % HW;
    dst[i] = src[(size_t)c * NS + (size_t)b * HW + hw];
}

// weight fp32 [M,K] -> fp16 [Mp,Kp] zero padded (row-major)
__global__ void k_w2h(const float* __restrict__ w, _Float16* __restrict__ o,
                      int M, int K, int Mp, int Kp) {
    int i = blockIdx.x * blockDim.x + threadIdx.x;
    if (i >= Mp * Kp) return;
    int m = i / Kp, k = i % Kp;
    float v = (m < M && k < K) ? w[m * K + k] : 0.f;
    o[i] = (_Float16)v;
}

// ---------------- LayerNorm over channels (+ optional skip add) ----------------
// x,add: [C, NS] fp32 ; out: pixel-major [NS, C] fp16
__global__ void k_ln_add(const float* __restrict__ x,
                         const float* __restrict__ add,
                         const float* __restrict__ lw, const float* __restrict__ lb,
                         _Float16* __restrict__ out) {
    int p = blockIdx.x * blockDim.x + threadIdx.x;
    float s = 0.f, ss = 0.f;
    for (int c = 0; c < C_DIM; ++c) {
        float v = x[(size_t)c * NS + p];
        s += v; ss += v * v;
    }
    float mu  = s * (1.f / C_DIM);
    float var = ss * (1.f / C_DIM) - mu * mu;
    float rs  = rsqrtf(var + EPS_LN);
    _Float16* orow = out + (size_t)p * C_DIM;
    v8h tmp;
    #pragma unroll 8
    for (int c = 0; c < C_DIM; ++c) {
        float v = (x[(size_t)c * NS + p] - mu) * rs * lw[c] + lb[c];
        if (add) v += add[(size_t)c * NS + p];
        tmp[c & 7] = (_Float16)v;
        if ((c & 7) == 7) *(v8h*)(orow + (c & ~7)) = tmp;
    }
}

// ---------------- WMMA GEMM: Out[m,n] = sum_k Wt[m,k]*X[n,k] ----------------
// Wt: fp16 [Mp,Kp] row-major.  X: fp16 pixel-major [NS,Kp].  Out: fp32 [Mp,NS].
// Block = 128 threads (4 waves). Block tile: 32(m) x 64(n); each wave owns a
// 16-pixel column group and computes two stacked 16x16 M-tiles sharing one B frag.
__global__ __launch_bounds__(128) void k_gemm_f16(
        const _Float16* __restrict__ Wt, const _Float16* __restrict__ X,
        float* __restrict__ Out, int Kpad, int Mreal) {
    const int tid  = threadIdx.x;
    const int wave = tid >> 5, lane = tid & 31;
    const int mbase = blockIdx.y * 32;
    const int ncol  = blockIdx.x * 64 + wave * 16 + (lane & 15);
    const int M     = lane & 15;
    const int kb    = (lane < 16) ? 0 : 8;
    const int kb2   = (lane < 16) ? 0 : 16;
    const _Float16* arow0 = Wt + (size_t)(mbase + M) * Kpad;
    const _Float16* arow1 = arow0 + (size_t)16 * Kpad;
    const _Float16* brow  = X + (size_t)ncol * Kpad;
    v8f c0 = {}, c1 = {};
    #pragma unroll 4
    for (int k0 = 0; k0 < Kpad; k0 += 32) {
        v16h bfrag = *(const v16h*)(brow + k0 + kb2);
        v8h lo0 = *(const v8h*)(arow0 + k0 + kb2);
        v8h hi0 = *(const v8h*)(arow0 + k0 + 8 + kb2);
        v8h lo1 = *(const v8h*)(arow1 + k0 + kb2);
        v8h hi1 = *(const v8h*)(arow1 + k0 + 8 + kb2);
        v16h a0, a1;
        #pragma unroll
        for (int j = 0; j < 8; ++j) {
            a0[j] = lo0[j]; a0[8 + j] = hi0[j];
            a1[j] = lo1[j]; a1[8 + j] = hi1[j];
        }
        c0 = __builtin_amdgcn_wmma_f32_16x16x32_f16(false, a0, false, bfrag,
                                                    (short)0, c0, false, false);
        c1 = __builtin_amdgcn_wmma_f32_16x16x32_f16(false, a1, false, bfrag,
                                                    (short)0, c1, false, false);
    }
    int mrow = 8 * (lane >> 4);
    #pragma unroll
    for (int r = 0; r < 8; ++r) {
        int m0 = mbase + mrow + r;
        if (m0 < Mreal)      Out[(size_t)m0 * NS + ncol]        = c0[r];
        int m1 = m0 + 16;
        if (m1 < Mreal)      Out[(size_t)m1 * NS + ncol]        = c1[r];
    }
}

// ---------------- depthwise 3x3, pad 1 (row-major in/out) ----------------
__global__ void k_dwconv3x3(const float* __restrict__ in, const float* __restrict__ w,
                            float* __restrict__ out) {
    int p = blockIdx.x * blockDim.x + threadIdx.x;   // < NS
    int c = blockIdx.y;
    int b = p >> 16;
    int hw = p & (HW - 1);
    int y = hw >> 8, x = hw & 255;
    const float* wr  = w + c * 9;
    const float* row = in + (size_t)c * NS + (size_t)b * HW;
    float acc = 0.f;
    #pragma unroll
    for (int ky = -1; ky <= 1; ++ky) {
        int yy = y + ky;
        if (yy < 0 || yy >= H_DIM) continue;
        #pragma unroll
        for (int kx = -1; kx <= 1; ++kx) {
            int xx = x + kx;
            if (xx < 0 || xx >= W_DIM) continue;
            acc += wr[(ky + 1) * 3 + (kx + 1)] * row[yy * W_DIM + xx];
        }
    }
    out[(size_t)c * NS + p] = acc;
}

// ---------------- L2 norm over pixels (per batch per row) ----------------
__global__ void k_sumsq(const float* __restrict__ src, float* __restrict__ norms) {
    __shared__ float sdata[256];
    int r = blockIdx.y;            // 0..191 (q rows 0..95, k rows 96..191)
    int b = blockIdx.z;            // 0..1
    const float* p = src + (size_t)r * NS + (size_t)b * HW + (size_t)blockIdx.x * 1024;
    float s = 0.f;
    for (int i = threadIdx.x; i < 1024; i += 256) { float v = p[i]; s += v * v; }
    sdata[threadIdx.x] = s;
    __syncthreads();
    for (int st = 128; st > 0; st >>= 1) {
        if ((int)threadIdx.x < st) sdata[threadIdx.x] += sdata[threadIdx.x + st];
        __syncthreads();
    }
    if (threadIdx.x == 0) atomicAdd(&norms[r * 2 + b], sdata[0]);
}

__global__ void k_l2scale(const float* __restrict__ src, const float* __restrict__ norms,
                          _Float16* __restrict__ dst) {    // dst row-major [96, NS]
    int p = blockIdx.x * blockDim.x + threadIdx.x;
    int c = blockIdx.y;            // 0..95
    int b = p >> 16;
    float n  = sqrtf(norms[c * 2 + b]);
    float sc = 1.f / fmaxf(n, 1e-12f);
    dst[(size_t)c * NS + p] = (_Float16)(src[(size_t)c * NS + p] * sc);
}

// ---------------- attn = q . k^T (sum over 65536 pixels) via WMMA ----------------
__global__ __launch_bounds__(32) void k_attn_qk(const _Float16* __restrict__ q,
                                                const _Float16* __restrict__ k,
                                                float* __restrict__ attn) {
    int lane = threadIdx.x;
    int bh = blockIdx.y;           // 0..11
    int b = bh / HEADS, h = bh % HEADS;
    size_t nbeg = (size_t)b * HW + (size_t)blockIdx.x * 512;
    int M   = lane & 15;
    int kb  = (lane < 16) ? 0 : 8;
    int kb2 = (lane < 16) ? 0 : 16;
    const _Float16* qrow = q + (size_t)(h * HD + M) * NS + nbeg;  // A row (c)
    const _Float16* krow = k + (size_t)(h * HD + M) * NS + nbeg;  // B col (d = lane&15)
    v8f c = {};
    #pragma unroll 4
    for (int n0 = 0; n0 < 512; n0 += 32) {
        v8h lo = *(const v8h*)(qrow + n0 + kb2);
        v8h hi = *(const v8h*)(qrow + n0 + 8 + kb2);
        v16h a;
        #pragma unroll
        for (int j = 0; j < 8; ++j) { a[j] = lo[j]; a[8 + j] = hi[j]; }
        v16h bfrag = *(const v16h*)(krow + n0 + kb2);
        c = __builtin_amdgcn_wmma_f32_16x16x32_f16(false, a, false, bfrag,
                                                   (short)0, c, false, false);
    }
    int mrow = 8 * (lane >> 4);
    int ncol = lane & 15;
    float* dst = attn + (size_t)bh * 256;
    #pragma unroll
    for (int r = 0; r < 8; ++r)
        atomicAdd(&dst[(mrow + r) * 16 + ncol], c[r]);
}

// ---------------- softmax over last dim of [B*H,16,16] with temperature ----------------
__global__ void k_softmax(float* __restrict__ attn, const float* __restrict__ temp) {
    int row = threadIdx.x;
    if (row >= B_DIM * HEADS * HD) return;
    int bh = row >> 4;
    float t = temp[bh % HEADS];
    float* p = attn + (size_t)bh * 256 + (row & 15) * 16;
    float v[16], mx = -1e30f;
    for (int i = 0; i < 16; ++i) { v[i] = p[i] * t; mx = fmaxf(mx, v[i]); }
    float s = 0.f;
    for (int i = 0; i < 16; ++i) { v[i] = expf(v[i] - mx); s += v[i]; }
    float inv = 1.f / s;
    for (int i = 0; i < 16; ++i) p[i] = v[i] * inv;
}

// ---------------- out = attn @ v ; out is pixel-major [NS, 96] fp16 ----------------
__global__ void k_attn_v(const float* __restrict__ attn, const float* __restrict__ v,
                         _Float16* __restrict__ out) {
    __shared__ float sA[256];
    int p = blockIdx.x * blockDim.x + threadIdx.x;
    int h = blockIdx.y;
    int b = p >> 16;               // whole block shares one b (256 | 65536)
    sA[threadIdx.x] = attn[(size_t)(b * HEADS + h) * 256 + threadIdx.x];
    __syncthreads();
    float acc[HD];
    #pragma unroll
    for (int i = 0; i < HD; ++i) acc[i] = 0.f;
    for (int d = 0; d < HD; ++d) {
        float vv = v[(size_t)(h * HD + d) * NS + p];
        #pragma unroll
        for (int cc = 0; cc < HD; ++cc) acc[cc] += sA[cc * 16 + d] * vv;
    }
    v8h lo, hi;
    #pragma unroll
    for (int cc = 0; cc < 8; ++cc) { lo[cc] = (_Float16)acc[cc]; hi[cc] = (_Float16)acc[8 + cc]; }
    _Float16* orow = out + (size_t)p * C_DIM + h * HD;
    *(v8h*)(orow)     = lo;
    *(v8h*)(orow + 8) = hi;
}

// ---------------- residual add ----------------
__global__ void k_add(float* __restrict__ x, const float* __restrict__ y) {
    size_t i = (size_t)blockIdx.x * blockDim.x + threadIdx.x;
    x[i] += y[i];
}

// ---------------- GELU(x1)*x2 gate : dw row-major [510,NS] -> g pixel-major [NS,256] ----------------
__global__ void k_gelu_gate(const float* __restrict__ dw, _Float16* __restrict__ g) {
    int p = blockIdx.x * blockDim.x + threadIdx.x;
    _Float16* grow = g + (size_t)p * FFN_KP;
    v8h tmp;
    #pragma unroll 8
    for (int r = 0; r < FFN_KP; ++r) {
        float o = 0.f;
        if (r < HID) {
            float x1 = dw[(size_t)r * NS + p];
            float x2 = dw[(size_t)(HID + r) * NS + p];
            float ge = 0.5f * x1 * (1.f + erff(x1 * 0.70710678118654752f));
            o = ge * x2;
        }
        tmp[r & 7] = (_Float16)o;
        if ((r & 7) == 7) *(v8h*)(grow + (r & ~7)) = tmp;
    }
}

// =======================================================================
extern "C" void kernel_launch(void* const* d_in, const int* in_sizes, int n_in,
                              void* d_out, int out_size, void* d_ws, size_t ws_size,
                              hipStream_t stream) {
    (void)in_sizes; (void)n_in; (void)out_size; (void)ws_size;
    const float* Fw      = (const float*)d_in[0];
    const float* F0c     = (const float*)d_in[1];
    const float* Kd      = (const float*)d_in[2];
    const float* ln1w    = (const float*)d_in[3];
    const float* ln1b    = (const float*)d_in[4];
    const float* a1_qkv  = (const float*)d_in[5];
    const float* a1_dw   = (const float*)d_in[6];
    const float* a1_proj = (const float*)d_in[7];
    const float* a1_temp = (const float*)d_in[8];
    const float* ln2w    = (const float*)d_in[9];
    const float* ln2b    = (const float*)d_in[10];
    const float* f1_in   = (const float*)d_in[11];
    const float* f1_dww  = (const float*)d_in[12];
    const float* f1_out  = (const float*)d_in[13];
    const float* ln3w    = (const float*)d_in[14];
    const float* ln3b    = (const float*)d_in[15];
    const float* a2_qkv  = (const float*)d_in[16];
    const float* a2_dw   = (const float*)d_in[17];
    const float* a2_proj = (const float*)d_in[18];
    const float* a2_temp = (const float*)d_in[19];
    const float* ln4w    = (const float*)d_in[20];
    const float* ln4b    = (const float*)d_in[21];
    const float* f2_in   = (const float*)d_in[22];
    const float* f2_dww  = (const float*)d_in[23];
    const float* f2_out  = (const float*)d_in[24];

    char* ws = (char*)d_ws;
    size_t off = 0;
    auto alloc = [&](size_t bytes) -> void* {
        void* p = ws + off;
        off = (off + bytes + 255) & ~(size_t)255;
        return p;
    };
    float*    xres  = (float*)   alloc((size_t)C_DIM * NS * 4);
    float*    f0cn  = (float*)   alloc((size_t)C_DIM * NS * 4);
    float*    kdcn  = (float*)   alloc((size_t)C_DIM * NS * 4);
    _Float16* x16   = (_Float16*)alloc((size_t)NS * C_DIM * 2);     // pixel-major [NS,96]
    _Float16* q16   = (_Float16*)alloc((size_t)C_DIM * NS * 2);     // row-major   [96,NS]
    _Float16* k16   = (_Float16*)alloc((size_t)C_DIM * NS * 2);     // row-major   [96,NS]
    _Float16* att16 = (_Float16*)alloc((size_t)NS * C_DIM * 2);     // pixel-major [NS,96]
    float*    gbuf  = (float*)   alloc((size_t)FFN_MP * NS * 4);
    float*    dbuf  = (float*)   alloc((size_t)FFN_MP * NS * 4);
    _Float16* g16   = (_Float16*)alloc((size_t)NS * FFN_KP * 2);    // pixel-major [NS,256]
    float*    norms = (float*)   alloc(192 * 2 * 4);
    float*    attnb = (float*)   alloc((size_t)B_DIM * HEADS * 256 * 4);
    _Float16* wq1 = (_Float16*)alloc(288 * 96 * 2);
    _Float16* wp1 = (_Float16*)alloc(96 * 96 * 2);
    _Float16* wi1 = (_Float16*)alloc((size_t)FFN_MP * 96 * 2);
    _Float16* wo1 = (_Float16*)alloc(96 * FFN_KP * 2);
    _Float16* wq2 = (_Float16*)alloc(288 * 96 * 2);
    _Float16* wp2 = (_Float16*)alloc(96 * 96 * 2);
    _Float16* wi2 = (_Float16*)alloc((size_t)FFN_MP * 96 * 2);
    _Float16* wo2 = (_Float16*)alloc(96 * FFN_KP * 2);

    const dim3 blk(256);
    const int NELEM_BLKS = (B_DIM * C_DIM * HW) / 256;   // 49152

    // ---- prep: layout + weight conversion ----
    k_cvt_nchw_to_cn<<<NELEM_BLKS, blk, 0, stream>>>(Fw,  xres);
    k_cvt_nchw_to_cn<<<NELEM_BLKS, blk, 0, stream>>>(F0c, f0cn);
    k_cvt_nchw_to_cn<<<NELEM_BLKS, blk, 0, stream>>>(Kd,  kdcn);
    auto w2h = [&](const float* w, _Float16* o, int M, int K, int Mp, int Kp) {
        int n = Mp * Kp;
        k_w2h<<<(n + 255) / 256, blk, 0, stream>>>(w, o, M, K, Mp, Kp);
    };
    w2h(a1_qkv,  wq1, 288, 96, 288, 96);
    w2h(a1_proj, wp1, 96,  96, 96,  96);
    w2h(f1_in,   wi1, FFN_M, 96, FFN_MP, 96);
    w2h(f1_out,  wo1, 96, HID, 96, FFN_KP);
    w2h(a2_qkv,  wq2, 288, 96, 288, 96);
    w2h(a2_proj, wp2, 96,  96, 96,  96);
    w2h(f2_in,   wi2, FFN_M, 96, FFN_MP, 96);
    w2h(f2_out,  wo2, 96, HID, 96, FFN_KP);

    auto attention = [&](const float* fcn, const _Float16* wqkv, const _Float16* wproj,
                         const float* dww, const float* temp,
                         const float* lnw_, const float* lnb_) {
        k_ln_add<<<NS / 256, blk, 0, stream>>>(xres, fcn, lnw_, lnb_, x16);
        k_gemm_f16<<<dim3(NS / 64, 288 / 32), dim3(128), 0, stream>>>(wqkv, x16, gbuf, 96, 288);
        k_dwconv3x3<<<dim3(NS / 256, 288), blk, 0, stream>>>(gbuf, dww, dbuf);
        hipMemsetAsync(norms, 0, 192 * 2 * 4, stream);
        k_sumsq<<<dim3(64, 192, 2), blk, 0, stream>>>(dbuf, norms);
        k_l2scale<<<dim3(NS / 256, 96), blk, 0, stream>>>(dbuf, norms, q16);
        k_l2scale<<<dim3(NS / 256, 96), blk, 0, stream>>>(dbuf + (size_t)96 * NS, norms + 192, k16);
        hipMemsetAsync(attnb, 0, (size_t)B_DIM * HEADS * 256 * 4, stream);
        k_attn_qk<<<dim3(HW / 512, B_DIM * HEADS), dim3(32), 0, stream>>>(q16, k16, attnb);
        k_softmax<<<1, 192, 0, stream>>>(attnb, temp);
        k_attn_v<<<dim3(NS / 256, HEADS), blk, 0, stream>>>(attnb, dbuf + (size_t)192 * NS, att16);
        k_gemm_f16<<<dim3(NS / 64, 96 / 32), dim3(128), 0, stream>>>(wproj, att16, gbuf, 96, 96);
        k_add<<<NELEM_BLKS, blk, 0, stream>>>(xres, gbuf);
    };

    auto ffn = [&](const _Float16* win, const _Float16* wout, const float* dww,
                   const float* lnw_, const float* lnb_) {
        k_ln_add<<<NS / 256, blk, 0, stream>>>(xres, nullptr, lnw_, lnb_, x16);
        k_gemm_f16<<<dim3(NS / 64, FFN_MP / 32), dim3(128), 0, stream>>>(win, x16, gbuf, 96, FFN_M);
        k_dwconv3x3<<<dim3(NS / 256, FFN_M), blk, 0, stream>>>(gbuf, dww, dbuf);
        k_gelu_gate<<<NS / 256, blk, 0, stream>>>(dbuf, g16);
        k_gemm_f16<<<dim3(NS / 64, 96 / 32), dim3(128), 0, stream>>>(wout, g16, gbuf, FFN_KP, 96);
        k_add<<<NELEM_BLKS, blk, 0, stream>>>(xres, gbuf);
    };

    attention(f0cn, wq1, wp1, a1_dw, a1_temp, ln1w, ln1b);
    ffn(wi1, wo1, f1_dww, ln2w, ln2b);
    attention(kdcn, wq2, wp2, a2_dw, a2_temp, ln3w, ln3b);
    ffn(wi2, wo2, f2_dww, ln4w, ln4b);

    k_cn_to_nchw<<<NELEM_BLKS, blk, 0, stream>>>(xres, (float*)d_out);
}
